// MultiHeadAttention_11433202942880
// MI455X (gfx1250) — matrix operand order repaired and checked by reading, not compiled, not run on hardware
//
#include <hip/hip_runtime.h>
#include <hip/hip_bf16.h>

typedef __attribute__((ext_vector_type(16))) _Float16 v16h;
typedef __attribute__((ext_vector_type(8)))  _Float16 v8h;
typedef __attribute__((ext_vector_type(8)))  float    v8f;
typedef __attribute__((ext_vector_type(4)))  float    v4f;

#define D_MODEL 1024
#define HEADS   16
#define DK      64
#define SEQ     2048
#define BATCH   2

__device__ __forceinline__ v8f wmma16(v16h a, v16h b, v8f c) {
  // D = A(16x32 f16) * B(32x16 f16) + C(16x16 f32)
  return __builtin_amdgcn_wmma_f32_16x16x32_f16(false, a, false, b, (short)0, c,
                                                false, false);
}

// A-fragment (M=16, K=32, f16): lane<16 -> halfs {kb+0..7, kb+16..23},
// lane>=16 -> halfs {kb+8..15, kb+24..31}. hi8 = (lane>=16)?8:0.
__device__ __forceinline__ v16h afrag_f16(const _Float16* row, int kb, int hi8) {
  const v8h lo = *reinterpret_cast<const v8h*>(row + kb + hi8);
  const v8h hh = *reinterpret_cast<const v8h*>(row + kb + 16 + hi8);
  v16h r;
#pragma unroll
  for (int i = 0; i < 8; ++i) { r[i] = lo[i]; r[8 + i] = hh[i]; }
  return r;
}

// B-fragment (K=32, N=16, f16): lane<16 -> col lane, K {kb+0..7, kb+16..23};
// lane>=16 -> col lane-16, K {kb+8..15, kb+24..31}. hi16 = (lane>=16)?16:0.
__device__ __forceinline__ v16h bfrag_f16(const _Float16* row, int kb, int hi16) {
  return afrag_f16(row, kb, hi16 >> 1);
}

// ---------------------------------------------------------------------------
// Kernel 0: one-pass f32 -> f16 conversion (8 elems / thread, b128 in/out)
// ---------------------------------------------------------------------------
__global__ __launch_bounds__(256) void cvt_f16_kernel(
    const float* __restrict__ src, _Float16* __restrict__ dst) {
  const size_t i = ((size_t)blockIdx.x * 256 + threadIdx.x) * 8;
  const v4f x0 = *reinterpret_cast<const v4f*>(src + i);
  const v4f x1 = *reinterpret_cast<const v4f*>(src + i + 4);
  v8h o;
#pragma unroll
  for (int j = 0; j < 4; ++j) {
    o[j]     = (_Float16)x0[j];
    o[4 + j] = (_Float16)x1[j];
  }
  *reinterpret_cast<v8h*>(dst + i) = o;
}

// ---------------------------------------------------------------------------
// Kernel 1: q = Q@WQ^T, k = K@WK^T, v = V@WV^T. One wave per 32x32 tile
// (2x2 accumulators; 4 WMMAs per 2 A-frags + 2 B-frags per K-chunk).
// q,k stored f16 [N,H,S,DK]; v stored f16 transposed [N,H,DV,S].
// ---------------------------------------------------------------------------
__global__ __launch_bounds__(256) void qkv_proj_kernel(
    const _Float16* __restrict__ Qf, const _Float16* __restrict__ Kf,
    const _Float16* __restrict__ Vf, const _Float16* __restrict__ WQf,
    const _Float16* __restrict__ WKf, const _Float16* __restrict__ WVf,
    _Float16* __restrict__ qf, _Float16* __restrict__ kf,
    _Float16* __restrict__ vT) {
  const int wid  = blockIdx.x * 8 + (threadIdx.x >> 5);
  const int lane = threadIdx.x & 31;
  const int l    = lane & 15;
  const int hi8  = (lane >> 4) * 8;
  const int hi16 = (lane >> 4) * 16;

  const int mat = wid >> 12;            // 4096 tiles per matrix
  const int t   = wid & 4095;
  const int rt  = t >> 5, ct = t & 31;  // 128 x 32 tiles of 32x32
  const int row0 = rt * 32, col0 = ct * 32;

  const _Float16* X;
  const _Float16* W;
  if (mat == 0)      { X = Qf; W = WQf; }
  else if (mat == 1) { X = Kf; W = WKf; }
  else               { X = Vf; W = WVf; }

  const _Float16* xrow0 = X + (size_t)(row0 + l) * D_MODEL;
  const _Float16* xrow1 = xrow0 + (size_t)16 * D_MODEL;
  const _Float16* wrow0 = W + (size_t)(col0 + l) * D_MODEL;
  const _Float16* wrow1 = wrow0 + (size_t)16 * D_MODEL;

  v8f acc[2][2] = {};
  for (int c = 0; c < D_MODEL / 32; ++c) {
    const int kb = c * 32;
    const v16h a0 = afrag_f16(xrow0, kb, hi8);
    const v16h a1 = afrag_f16(xrow1, kb, hi8);
    const v16h b0 = bfrag_f16(wrow0, kb, hi16);
    const v16h b1 = bfrag_f16(wrow1, kb, hi16);
    acc[0][0] = wmma16(a0, b0, acc[0][0]);
    acc[0][1] = wmma16(a0, b1, acc[0][1]);
    acc[1][0] = wmma16(a1, b0, acc[1][0]);
    acc[1][1] = wmma16(a1, b1, acc[1][1]);
  }

  const int n_idx = row0 >> 11;                 // /SEQ (rows 0..31 same n)
  const int seq0  = (row0 & (SEQ - 1)) + hi8;

  if (mat < 2) {
    _Float16* dstm = (mat == 0) ? qf : kf;
#pragma unroll
    for (int ri = 0; ri < 2; ++ri) {
#pragma unroll
      for (int ci = 0; ci < 2; ++ci) {
        const int colb = col0 + ci * 16;
        const int h = colb >> 6, dk0 = colb & 63;
        _Float16* base = dstm +
            (((size_t)(n_idx * HEADS + h)) * SEQ + seq0 + ri * 16) * DK +
            dk0 + l;
#pragma unroll
        for (int r = 0; r < 8; ++r)
          base[(size_t)r * DK] = (_Float16)acc[ri][ci][r];
      }
    }
  } else {
#pragma unroll
    for (int ri = 0; ri < 2; ++ri) {
#pragma unroll
      for (int ci = 0; ci < 2; ++ci) {
        const int colb = col0 + ci * 16;
        const int h = colb >> 6, dk0 = colb & 63;
        _Float16* base = vT +
            (((size_t)(n_idx * HEADS + h)) * DK + dk0 + l) * SEQ + seq0 +
            ri * 16;
        v8h pk;
#pragma unroll
        for (int r = 0; r < 8; ++r) pk[r] = (_Float16)acc[ri][ci][r];
        *reinterpret_cast<v8h*>(base) = pk;
      }
    }
  }
}

// ---------------------------------------------------------------------------
// Kernel 2: causal flash attention. One wave per (n, h, 16-query tile).
// Computes scores^T so each lane owns one query column; the scores^T C-frag
// is exactly the P A-frag layout for P*V (no LDS / cross-lane transpose).
// ---------------------------------------------------------------------------
__global__ __launch_bounds__(256) void attn_kernel(
    const _Float16* __restrict__ qf, const _Float16* __restrict__ kf,
    const _Float16* __restrict__ vT, _Float16* __restrict__ attn) {
  const int wid  = blockIdx.x * 8 + (threadIdx.x >> 5);   // 0..4095
  const int lane = threadIdx.x & 31;
  const int l    = lane & 15;
  const int hi   = lane >> 4;
  const int rbase = hi * 8;

  const int n  = wid >> 11;
  const int h  = (wid >> 7) & 15;
  const int qt = wid & 127;
  const int q0 = qt * 16;

  const size_t headoff = (size_t)(n * HEADS + h) * SEQ * DK;
  const _Float16* qh = qf + headoff;
  const _Float16* kh = kf + headoff;
  const _Float16* vh = vT + headoff;   // [DV][S] for this head

  // Q as B-fragments (K=dk, N=query): lane column = query q0+l.
  const _Float16* qrow = qh + (size_t)(q0 + l) * DK;
  const v16h qb0 = bfrag_f16(qrow, 0, hi * 16);    // dk 0..31
  const v16h qb1 = bfrag_f16(qrow, 32, hi * 16);   // dk 32..63

  float m = -INFINITY, lsum = 0.f;
  v8f o0 = {}, o1 = {}, o2 = {}, o3 = {};
  const int qidx = q0 + l;
  const int kend = q0 + 16;

  for (int k0 = 0; k0 < kend; k0 += 32) {
    const _Float16* kr0 = kh + (size_t)(k0 + l) * DK;
    const _Float16* kr1 = kr0 + 16 * DK;
    __builtin_prefetch(kr0 + 32 * DK, 0, 1);   // next key tile

    v8f z = {};
    v16h ka = afrag_f16(kr0, 0, hi * 8);
    v8f st0 = wmma16(ka, qb0, z);
    ka = afrag_f16(kr0, 32, hi * 8);
    st0 = wmma16(ka, qb1, st0);
    ka = afrag_f16(kr1, 0, hi * 8);
    v8f st1 = wmma16(ka, qb0, z);
    ka = afrag_f16(kr1, 32, hi * 8);
    st1 = wmma16(ka, qb1, st1);

    // scale + causal mask (lane = query column; VGPR r = key row)
    const bool masked = (k0 + 31) > q0;
    const int kb0 = k0 + rbase;
    float sc[16];
#pragma unroll
    for (int r = 0; r < 8; ++r) {
      float s0 = st0[r] * 0.125f;
      float s1 = st1[r] * 0.125f;
      if (masked) {
        if (kb0 + r > qidx)      s0 = -INFINITY;
        if (kb0 + 16 + r > qidx) s1 = -INFINITY;
      }
      sc[r] = s0;
      sc[8 + r] = s1;
    }

    // online softmax: in-lane reduce + combine the two key-row halves
    float mloc = sc[0];
#pragma unroll
    for (int i = 1; i < 16; ++i) mloc = fmaxf(mloc, sc[i]);
    mloc = fmaxf(mloc, __shfl_xor(mloc, 16, 32));
    const float mnew = fmaxf(m, mloc);
    const float corr = __expf(m - mnew);

    float psum = 0.f;
    v16h pf;   // == P A-fragment (M=query, K=key 0..31) by construction
#pragma unroll
    for (int i = 0; i < 16; ++i) {
      const float p = __expf(sc[i] - mnew);
      pf[i] = (_Float16)p;
      psum += p;
    }
    psum += __shfl_xor(psum, 16, 32);
    lsum = lsum * corr + psum;
    m = mnew;

    // rescale O accumulators: broadcast per-query-row factors
    float cr[8];
#pragma unroll
    for (int r = 0; r < 8; ++r) cr[r] = __shfl(corr, rbase + r, 32);
#pragma unroll
    for (int r = 0; r < 8; ++r) {
      o0[r] *= cr[r]; o1[r] *= cr[r]; o2[r] *= cr[r]; o3[r] *= cr[r];
    }

    // P * V : B-fragments straight from transposed V
    const _Float16* vb = vh + (size_t)l * SEQ + k0;
    __builtin_prefetch(vb + 32, 0, 1);
    o0 = wmma16(pf, bfrag_f16(vb, 0, hi * 16), o0);
    o1 = wmma16(pf, bfrag_f16(vb + 16 * SEQ, 0, hi * 16), o1);
    o2 = wmma16(pf, bfrag_f16(vb + 32 * SEQ, 0, hi * 16), o2);
    o3 = wmma16(pf, bfrag_f16(vb + 48 * SEQ, 0, hi * 16), o3);
  }

  const float linv = 1.0f / lsum;
  float li[8];
#pragma unroll
  for (int r = 0; r < 8; ++r) li[r] = __shfl(linv, rbase + r, 32);

  _Float16* arow =
      attn + ((size_t)n * SEQ + q0 + rbase) * D_MODEL + h * DK + l;
#pragma unroll
  for (int r = 0; r < 8; ++r) {
    arow[(size_t)r * D_MODEL + 0]  = (_Float16)(o0[r] * li[r]);
    arow[(size_t)r * D_MODEL + 16] = (_Float16)(o1[r] * li[r]);
    arow[(size_t)r * D_MODEL + 32] = (_Float16)(o2[r] * li[r]);
    arow[(size_t)r * D_MODEL + 48] = (_Float16)(o3[r] * li[r]);
  }
}

// ---------------------------------------------------------------------------
// Kernel 3: out = attn @ WO^T + bO  (f32 output), 32x32 tile per wave
// ---------------------------------------------------------------------------
__global__ __launch_bounds__(256) void out_proj_kernel(
    const _Float16* __restrict__ attn, const _Float16* __restrict__ WOf,
    const float* __restrict__ bO, float* __restrict__ out) {
  const int wid  = blockIdx.x * 8 + (threadIdx.x >> 5);   // 0..4095
  const int lane = threadIdx.x & 31;
  const int l    = lane & 15;
  const int hi8  = (lane >> 4) * 8;
  const int hi16 = (lane >> 4) * 16;
  const int rt   = wid >> 5, ct = wid & 31;
  const int row0 = rt * 32, col0 = ct * 32;

  const _Float16* arow0 = attn + (size_t)(row0 + l) * D_MODEL;
  const _Float16* arow1 = arow0 + (size_t)16 * D_MODEL;
  const _Float16* wrow0 = WOf + (size_t)(col0 + l) * D_MODEL;
  const _Float16* wrow1 = wrow0 + (size_t)16 * D_MODEL;

  v8f acc[2][2] = {};
  for (int c = 0; c < D_MODEL / 32; ++c) {
    const int kb = c * 32;
    const v16h a0 = afrag_f16(arow0, kb, hi8);
    const v16h a1 = afrag_f16(arow1, kb, hi8);
    const v16h b0 = bfrag_f16(wrow0, kb, hi16);
    const v16h b1 = bfrag_f16(wrow1, kb, hi16);
    acc[0][0] = wmma16(a0, b0, acc[0][0]);
    acc[0][1] = wmma16(a0, b1, acc[0][1]);
    acc[1][0] = wmma16(a1, b0, acc[1][0]);
    acc[1][1] = wmma16(a1, b1, acc[1][1]);
  }

  const float bias0 = bO[col0 + l];
  const float bias1 = bO[col0 + 16 + l];
#pragma unroll
  for (int ri = 0; ri < 2; ++ri) {
    float* ob0 = out + (size_t)(row0 + ri * 16 + hi8) * D_MODEL + col0 + l;
#pragma unroll
    for (int r = 0; r < 8; ++r) {
      ob0[(size_t)r * D_MODEL]      = acc[ri][0][r] + bias0;
      ob0[(size_t)r * D_MODEL + 16] = acc[ri][1][r] + bias1;
    }
  }
}

// ---------------------------------------------------------------------------
extern "C" void kernel_launch(void* const* d_in, const int* in_sizes, int n_in,
                              void* d_out, int out_size, void* d_ws,
                              size_t ws_size, hipStream_t stream) {
  (void)in_sizes; (void)n_in; (void)out_size; (void)ws_size;
  const float* Q  = (const float*)d_in[0];
  const float* K  = (const float*)d_in[1];
  const float* V  = (const float*)d_in[2];
  const float* WQ = (const float*)d_in[3];
  const float* WK = (const float*)d_in[4];
  const float* WV = (const float*)d_in[5];
  const float* WO = (const float*)d_in[6];
  const float* bO = (const float*)d_in[7];
  float* out = (float*)d_out;

  const size_t NSD = (size_t)BATCH * SEQ * D_MODEL;  // 4,194,304
  const size_t DD  = (size_t)D_MODEL * D_MODEL;      // 1,048,576
  _Float16* ws   = (_Float16*)d_ws;
  _Float16* Qf   = ws;
  _Float16* Kf   = Qf + NSD;
  _Float16* Vf   = Kf + NSD;
  _Float16* WQf  = Vf + NSD;
  _Float16* WKf  = WQf + DD;
  _Float16* WVf  = WKf + DD;
  _Float16* WOf  = WVf + DD;
  _Float16* qf   = WOf + DD;
  _Float16* kf   = qf + NSD;
  _Float16* vT   = kf + NSD;
  _Float16* attn = vT + NSD;           // total 7*NSD + 4*DD = 64 MB

  // one-pass conversions (8 elems/thread)
  cvt_f16_kernel<<<dim3(NSD / (256 * 8)), dim3(256), 0, stream>>>(Q, Qf);
  cvt_f16_kernel<<<dim3(NSD / (256 * 8)), dim3(256), 0, stream>>>(K, Kf);
  cvt_f16_kernel<<<dim3(NSD / (256 * 8)), dim3(256), 0, stream>>>(V, Vf);
  cvt_f16_kernel<<<dim3(DD / (256 * 8)), dim3(256), 0, stream>>>(WQ, WQf);
  cvt_f16_kernel<<<dim3(DD / (256 * 8)), dim3(256), 0, stream>>>(WK, WKf);
  cvt_f16_kernel<<<dim3(DD / (256 * 8)), dim3(256), 0, stream>>>(WV, WVf);
  cvt_f16_kernel<<<dim3(DD / (256 * 8)), dim3(256), 0, stream>>>(WO, WOf);

  // 3 matrices * 128 * 32 tiles of 32x32 = 12288 waves, 8 waves/block
  qkv_proj_kernel<<<dim3(1536), dim3(256), 0, stream>>>(Qf, Kf, Vf, WQf, WKf,
                                                        WVf, qf, kf, vT);
  // 2 * 16 * 128 = 4096 waves
  attn_kernel<<<dim3(512), dim3(256), 0, stream>>>(qf, kf, vT, attn);
  // 128 * 32 tiles of 32x32 = 4096 waves
  out_proj_kernel<<<dim3(512), dim3(256), 0, stream>>>(attn, WOf, bO, out);
}
